// GAT_gate_74929999446928
// MI455X (gfx1250) — matrix operand order repaired and checked by reading, not compiled, not run on hardware
//
#include <hip/hip_runtime.h>
#include <stdint.h>
#include <math.h>

#define Bb 32
#define Nn 1024
#define Dd 128
#define NEGV (-9e15f)

typedef __attribute__((ext_vector_type(16))) _Float16 v16h;
typedef __attribute__((ext_vector_type(8)))  float    v8f;

// ---- WMMA helpers (16x16x32 f16 -> f32), wave32 ----

__device__ __forceinline__ v8f wmma32(v16h a, v16h b, v8f c) {
  return __builtin_amdgcn_wmma_f32_16x16x32_f16(
      /*neg_a=*/false, a, /*neg_b=*/false, b,
      /*c_mod=*/(short)0, c, /*reuse_a=*/false, /*reuse_b=*/false);
}

// A fragment: 16x32 f16, src = tile origin in row-major array, leading dim ld.
// Lanes 0-15 hold row M=lane, K=0..15; lanes 16-31 hold row M=lane-16,
// K=16..31 (2 halfs per VGPR).
__device__ __forceinline__ v16h frag_a(const _Float16* __restrict__ src, int ld) {
  int lane = threadIdx.x & 31;
  int row  = lane & 15;
  int kb   = (lane < 16) ? 0 : 16;
  const _Float16* p = src + row * ld + kb;
  v16h a;
#pragma unroll
  for (int j = 0; j < 4; ++j) { a[2*j]   = p[2*j];      a[2*j+1]   = p[2*j+1]; }
#pragma unroll
  for (int j = 0; j < 4; ++j) { a[8+2*j] = p[8 + 2*j]; a[8+2*j+1] = p[8 + 2*j + 1]; }
  return a;
}

// B fragment (direct): B[k][n] = src[k*ld + n]. Lanes 0-15: col n=lane, K=0..15;
// lanes 16-31: col n=lane-16, K=16..31 (contiguous K per lane half).
__device__ __forceinline__ v16h frag_b(const _Float16* __restrict__ src, int ld) {
  int lane = threadIdx.x & 31;
  int col  = lane & 15;
  int kb   = (lane < 16) ? 0 : 16;
  v16h b;
#pragma unroll
  for (int i = 0; i < 16; ++i) b[i] = src[(kb + i) * ld + col];
  return b;
}

// B fragment (transposed source): B[k][n] = src[n*ld + k]. Each lane reads 16
// contiguous halfs from one source row -> vectorizable b128 loads.
__device__ __forceinline__ v16h frag_bT(const _Float16* __restrict__ src, int ld) {
  int lane = threadIdx.x & 31;
  int col  = lane & 15;
  int kb   = (lane < 16) ? 0 : 16;
  const _Float16* p = src + col * ld + kb;
  v16h b;
#pragma unroll
  for (int i = 0; i < 16; ++i) b[i] = p[i];
  return b;
}

// ---- Kernel 1: f32 -> f16 downconvert of x, W_w, A ----
__global__ void k_convert(const float* __restrict__ x, const float* __restrict__ W,
                          const float* __restrict__ A, _Float16* __restrict__ xh,
                          _Float16* __restrict__ Wh, _Float16* __restrict__ Ah) {
  int i = blockIdx.x * 256 + threadIdx.x;
  if (i < Bb * Nn * Dd) xh[i] = (_Float16)x[i];
  if (i < Dd * Dd) { Wh[i] = (_Float16)W[i]; Ah[i] = (_Float16)A[i]; }
}

// ---- Kernel 2: h = x@W^T + b, then hA = h@A, fused per 32-row strip ----
__global__ void k_h_hA(const _Float16* __restrict__ xh, const _Float16* __restrict__ Wh,
                       const _Float16* __restrict__ Ah, const float* __restrict__ Wb,
                       _Float16* __restrict__ h16, _Float16* __restrict__ hA16) {
  __shared__ _Float16 sH[32 * Dd];
  int b     = blockIdx.x >> 5;
  int nbase = (blockIdx.x & 31) * 32;
  int wv    = threadIdx.x >> 5;
  int lane  = threadIdx.x & 31;
  int colb  = wv * 16;
  int col   = lane & 15;
  int roff  = (lane < 16) ? 0 : 8;

  float bias = Wb[colb + col];
  v8f acc[2];
#pragma unroll
  for (int m = 0; m < 2; ++m) {
    v8f c = {};
#pragma unroll
    for (int k = 0; k < 4; ++k) {
      v16h a  = frag_a(xh + ((size_t)(b * Nn + nbase + m * 16)) * Dd + k * 32, Dd);
      v16h bw = frag_bT(Wh + (size_t)colb * Dd + k * 32, Dd);  // B = W^T
      c = wmma32(a, bw, c);
    }
    acc[m] = c;
  }
#pragma unroll
  for (int m = 0; m < 2; ++m) {
#pragma unroll
    for (int r = 0; r < 8; ++r) {
      int lrow = m * 16 + r + roff;
      _Float16 hv = (_Float16)(acc[m][r] + bias);
      sH[lrow * Dd + colb + col] = hv;
      h16[((size_t)(b * Nn + nbase + lrow)) * Dd + colb + col] = hv;
    }
  }
  __syncthreads();
#pragma unroll
  for (int m = 0; m < 2; ++m) {
    v8f c = {};
#pragma unroll
    for (int k = 0; k < 4; ++k) {
      v16h a  = frag_a(sH + (m * 16) * Dd + k * 32, Dd);
      v16h ba = frag_b(Ah + (size_t)(k * 32) * Dd + colb, Dd);  // B = A direct
      c = wmma32(a, ba, c);
    }
#pragma unroll
    for (int r = 0; r < 8; ++r) {
      int lrow = m * 16 + r + roff;
      hA16[((size_t)(b * Nn + nbase + lrow)) * Dd + colb + col] = (_Float16)c[r];
    }
  }
}

// ---- Kernel 3: scores = mask(e + e^T), e_sym[I,J] = hA_I h_J^T + h_I hA_J^T ----
__global__ void k_scores(const _Float16* __restrict__ h16, const _Float16* __restrict__ hA16,
                         const float* __restrict__ adj, float* __restrict__ sc) {
  int b   = blockIdx.x >> 9;
  int rem = blockIdx.x & 511;
  int ib  = (rem >> 4) * 32;
  int jb  = (rem & 15) * 64;
  int wv   = threadIdx.x >> 5;
  int lane = threadIdx.x & 31;
  int it = ib + (wv >> 2) * 16;
  int jt = jb + (wv & 3) * 16;
  int roff = (lane < 16) ? 0 : 8;
  int mcol = jt + (lane & 15);
  // pull the adjacency lines we'll mask with into cache early
  __builtin_prefetch(&adj[((size_t)(b * Nn + it + roff)) * Nn + mcol], 0, 1);
  v8f acc = {};
#pragma unroll
  for (int k = 0; k < 4; ++k) {
    v16h aA = frag_a (hA16 + ((size_t)(b * Nn + it)) * Dd + k * 32, Dd);
    v16h bA = frag_bT(h16  + ((size_t)(b * Nn + jt)) * Dd + k * 32, Dd);
    acc = wmma32(aA, bA, acc);
    v16h aB = frag_a (h16  + ((size_t)(b * Nn + it)) * Dd + k * 32, Dd);
    v16h bB = frag_bT(hA16 + ((size_t)(b * Nn + jt)) * Dd + k * 32, Dd);
    acc = wmma32(aB, bB, acc);
  }
#pragma unroll
  for (int r = 0; r < 8; ++r) {
    int n = it + r + roff;
    size_t idx = ((size_t)(b * Nn + n)) * Nn + mcol;
    sc[idx] = (adj[idx] > 0.f) ? acc[r] : NEGV;
  }
}

// ---- Kernel 4: online softmax stats over axis=1 (per column m) ----
__global__ void k_stats(const float* __restrict__ sc, float* __restrict__ cm,
                        float* __restrict__ cs) {
  __shared__ float sM[4][64];
  __shared__ float sS[4][64];
  int b  = blockIdx.x >> 4;
  int cb = (blockIdx.x & 15) * 64;
  int tm = threadIdx.x & 63;
  int tn = threadIdx.x >> 6;
  int m  = cb + tm;
  float mx = -3.0e38f, s = 0.f;
  for (int n = tn; n < Nn; n += 4) {
    // streaming-bound pass: software-prefetch 8 strides (32 rows) ahead
    if (n + 32 < Nn)
      __builtin_prefetch(&sc[((size_t)(b * Nn + n + 32)) * Nn + m], 0, 1);
    float v  = sc[((size_t)(b * Nn + n)) * Nn + m];
    float nm = fmaxf(mx, v);
    s  = s * __expf(mx - nm) + __expf(v - nm);
    mx = nm;
  }
  sM[tn][tm] = mx; sS[tn][tm] = s;
  __syncthreads();
  if (tn == 0) {
#pragma unroll
    for (int i = 1; i < 4; ++i) {
      float m2 = sM[i][tm], s2 = sS[i][tm];
      float nm = fmaxf(mx, m2);
      s  = s * __expf(mx - nm) + s2 * __expf(m2 - nm);
      mx = nm;
    }
    cm[b * Nn + m] = mx;
    cs[b * Nn + m] = s;
  }
}

// ---- Kernel 5: h' = relu(attn @ h); coeff = sigmoid([x,h']@gw + gb); out ----
// Double-buffered LDS pipeline: h tiles fetched with GLOBAL_LOAD_ASYNC_TO_LDS
// (ASYNCcnt), overlapped with attention staging + WMMA on the current tile.
__global__ void k_out(const float* __restrict__ x, const float* __restrict__ adj,
                      const float* __restrict__ sc, const float* __restrict__ cm,
                      const float* __restrict__ cs, const _Float16* __restrict__ h16,
                      const float* __restrict__ gw, const float* __restrict__ gb,
                      float* __restrict__ out) {
  __shared__ _Float16 sAttn[2][32 * 32];
  __shared__ _Float16 sHt[2][32 * Dd];
  __shared__ float    sHp[32 * Dd];
  __shared__ float    sRed[32][8];
  __shared__ float    sCoeff[32];
  int b     = blockIdx.x >> 5;
  int ibase = (blockIdx.x & 31) * 32;
  int tid   = threadIdx.x;
  int wv = tid >> 5, lane = tid & 31;
  int colb = wv * 16, col = lane & 15, roff = (lane < 16) ? 0 : 8;

  const _Float16* hbase = h16 + (size_t)b * Nn * Dd;

  // issue the async global->LDS copy of one 32x128 f16 h tile (8KB = 512 x b128;
  // 2 async instructions per wave per tile)
  auto issue_h = [&](int buf, int j0) {
    uint32_t lds0 = (uint32_t)(uintptr_t)(&sHt[buf][0]);
    uint64_t g0   = (uint64_t)(uintptr_t)(hbase + (size_t)j0 * Dd);
#pragma unroll
    for (int it = 0; it < 2; ++it) {
      int idx = tid + it * 256;
      uint32_t l = lds0 + (uint32_t)idx * 16u;
      uint64_t g = g0 + (uint64_t)idx * 16u;
      asm volatile("global_load_async_to_lds_b128 %0, %1, off"
                   :: "v"(l), "v"(g) : "memory");
    }
  };

  issue_h(0, 0);
  v8f acc0 = {}, acc1 = {};
  for (int step = 0; step < Nn / 32; ++step) {
    int cur = step & 1;
    int j0  = step * 32;
    // stage normalized attention tile (f32 scores -> f16) into current buffer
    for (int idx = tid; idx < 1024; idx += 256) {
      int li = idx >> 5, lj = idx & 31;
      int n = ibase + li, jc = j0 + lj;
      size_t sidx = ((size_t)(b * Nn + n)) * Nn + jc;
      float a = adj[sidx];
      float v = __expf(sc[sidx] - cm[b * Nn + jc]) / cs[b * Nn + jc];
      sAttn[cur][li * 32 + lj] = (_Float16)(a * v);
    }
    // prefetch next h tile into the other buffer, then retire current tile
    if (step < Nn / 32 - 1) {
      issue_h(cur ^ 1, j0 + 32);
      asm volatile("s_wait_asynccnt 0x2" ::: "memory");
    } else {
      asm volatile("s_wait_asynccnt 0x0" ::: "memory");
    }
    __syncthreads();
    v16h bfr = frag_b(&sHt[cur][colb], Dd);
    acc0 = wmma32(frag_a(&sAttn[cur][0], 32), bfr, acc0);
    acc1 = wmma32(frag_a(&sAttn[cur][16 * 32], 32), bfr, acc1);
    __syncthreads();
  }
  // ReLU -> LDS (f32)
#pragma unroll
  for (int r = 0; r < 8; ++r) {
    sHp[(r + roff) * Dd + colb + col]      = fmaxf(acc0[r], 0.f);
    sHp[(16 + r + roff) * Dd + colb + col] = fmaxf(acc1[r], 0.f);
  }
  __syncthreads();
  // gate: 32 rows x 8 partial reducers of 16 cols each
  int li = tid >> 3, t = tid & 7;
  const float* xrow = x + ((size_t)(b * Nn + ibase + li)) * Dd;
  float p = 0.f;
#pragma unroll
  for (int c = t * 16; c < t * 16 + 16; ++c)
    p += xrow[c] * gw[c] + sHp[li * Dd + c] * gw[Dd + c];
  sRed[li][t] = p;
  __syncthreads();
  if (t == 0) {
    float z = gb[0];
#pragma unroll
    for (int i = 0; i < 8; ++i) z += sRed[li][i];
    sCoeff[li] = 1.f / (1.f + __expf(-z));
  }
  __syncthreads();
  for (int idx = tid; idx < 32 * Dd; idx += 256) {
    int r = idx >> 7, c = idx & 127;
    float cf = sCoeff[r];
    size_t gi = ((size_t)(b * Nn + ibase + r)) * Dd + c;
    out[gi] = cf * x[gi] + (1.f - cf) * sHp[r * Dd + c];
  }
}

extern "C" void kernel_launch(void* const* d_in, const int* in_sizes, int n_in,
                              void* d_out, int out_size, void* d_ws, size_t ws_size,
                              hipStream_t stream) {
  (void)in_sizes; (void)n_in; (void)out_size; (void)ws_size;
  const float* x   = (const float*)d_in[0];
  const float* adj = (const float*)d_in[1];
  const float* Ww  = (const float*)d_in[2];
  const float* Wb  = (const float*)d_in[3];
  const float* A   = (const float*)d_in[4];
  const float* gw  = (const float*)d_in[5];
  const float* gb  = (const float*)d_in[6];
  float* out = (float*)d_out;

  char* ws = (char*)d_ws;
  size_t off = 0;
  _Float16* xh   = (_Float16*)(ws + off); off += (size_t)Bb * Nn * Dd * 2;
  _Float16* Wh   = (_Float16*)(ws + off); off += (size_t)Dd * Dd * 2;
  _Float16* Ah   = (_Float16*)(ws + off); off += (size_t)Dd * Dd * 2;
  _Float16* h16  = (_Float16*)(ws + off); off += (size_t)Bb * Nn * Dd * 2;
  _Float16* hA16 = (_Float16*)(ws + off); off += (size_t)Bb * Nn * Dd * 2;
  float*    sc   = (float*)(ws + off);    off += (size_t)Bb * Nn * Nn * 4;
  float*    cm   = (float*)(ws + off);    off += (size_t)Bb * Nn * 4;
  float*    cs   = (float*)(ws + off);    off += (size_t)Bb * Nn * 4;

  k_convert<<<(Bb * Nn * Dd) / 256, 256, 0, stream>>>(x, Ww, A, xh, Wh, Ah);
  k_h_hA  <<<Bb * Nn / 32, 256, 0, stream>>>(xh, Wh, Ah, Wb, h16, hA16);
  k_scores<<<Bb * (Nn / 32) * (Nn / 64), 256, 0, stream>>>(h16, hA16, adj, sc);
  k_stats <<<Bb * (Nn / 64), 256, 0, stream>>>(sc, cm, cs);
  k_out   <<<Bb * Nn / 32, 256, 0, stream>>>(x, adj, sc, cm, cs, h16, gw, gb, out);
}
